// Decoder3_7928509629209
// MI455X (gfx1250) — matrix operand order repaired and checked by reading, not compiled, not run on hardware
//
#include <hip/hip_runtime.h>
#include <math.h>

// ---------------- problem constants ----------------
#define BB     4096
#define TDEC   10
#define TENC   100
#define HH     256
#define EE     52
#define NCLS   11
#define G3     768          // 3*H
#define KIH    128          // 104 padded to 128
#define KHH    256
#define BTILE  32           // batch rows per block

// workspace offsets (bytes)
#define OFF_A     0                         // a = pool@W_pool.T + b_reg : B x 16 f32
#define OFF_H0    (OFF_A  + BB*16*4)        // h double buffer 0
#define OFF_H1    (OFF_H0 + BB*HH*4)        // h double buffer 1
#define OFF_WIH   (OFF_H1 + BB*HH*4)        // bf16 W_ih  768x128
#define OFF_WHH   (OFF_WIH + G3*KIH*2)      // bf16 W_hh  768x256
#define OFF_WRG   (OFF_WHH + G3*KHH*2)      // bf16 W_reg 16x256

typedef __attribute__((ext_vector_type(16))) __bf16 v16bf;
typedef __attribute__((ext_vector_type(8)))  float  v8f;

union BFrag { v16bf v; uint4 q[2]; };
union FAcc  { v8f   v; float f[8]; };

__device__ __constant__ int kLabelCodes[NCLS] = {1, 2, 6, 13, 25, 33, 174, 39, 42, 44, 173};

__device__ inline unsigned short f2bf(float f) {
    unsigned int u = __float_as_uint(f);
    unsigned int r = u + 0x7FFFu + ((u >> 16) & 1u);   // round-to-nearest-even
    return (unsigned short)(r >> 16);
}

__device__ inline float sigmoidf_(float x) { return 1.0f / (1.0f + __expf(-x)); }

__device__ inline v8f wmma_bf16(v16bf a, v16bf b, v8f c) {
    return __builtin_amdgcn_wmma_f32_16x16x32_bf16(false, a, false, b, (short)0, c, false, false);
}

// A-fragment 16x32 bf16 from LDS, row-major [rows][ldk]
// lane L: M = L%16 ; K = kb + (L/16)*8 + {0..7}  and  kb + 16 + (L/16)*8 + {0..7}
__device__ inline v16bf load_a_frag(const unsigned short* base, int ldk, int m0, int kb, int lane) {
    int m  = m0 + (lane & 15);
    int kg = (lane >> 4) * 8;
    const unsigned short* p = base + m * ldk + kb + kg;
    BFrag f;
    f.q[0] = *(const uint4*)(p);
    f.q[1] = *(const uint4*)(p + 16);
    return f.v;
}

// B-fragment 32x16 bf16 from global weight stored row-major [N][ldk] (i.e. W, so B = W^T)
// lane L: N = n0 + L%16 ; K = kb + (L/16)*8 + {0..7}  and  kb + 16 + (L/16)*8 + {0..7}
__device__ inline v16bf load_b_frag(const unsigned short* W, int ldk, int n0, int kb, int lane) {
    int n = n0 + (lane & 15);
    int k = kb + (lane >> 4) * 8;
    const unsigned short* p = W + n * ldk + k;
    BFrag f;
    f.q[0] = *(const uint4*)(p);
    f.q[1] = *(const uint4*)(p + 16);
    return f.v;
}

// ---------------- one-time: f32 -> bf16 weight conversion (with padding) ----------------
__global__ void convert_weights(const float* __restrict__ W_ih, const float* __restrict__ W_hh,
                                const float* __restrict__ W_reg,
                                unsigned short* __restrict__ Wih, unsigned short* __restrict__ Whh,
                                unsigned short* __restrict__ Wrg) {
    const int NIH = G3 * KIH, NHH = G3 * KHH, NRG = 16 * KHH;
    int i = blockIdx.x * 256 + threadIdx.x;
    if (i < NIH) {
        int n = i / KIH, k = i % KIH;
        Wih[i] = f2bf(k < 104 ? W_ih[n * 104 + k] : 0.0f);
    } else if (i < NIH + NHH) {
        int j = i - NIH;
        Whh[j] = f2bf(W_hh[j]);
    } else if (i < NIH + NHH + NRG) {
        int j = i - NIH - NHH;
        int n = j / KHH, k = j % KHH;
        Wrg[j] = f2bf(n < NCLS ? W_reg[n * KHH + k] : 0.0f);
    }
}

// ---------------- pooling: pool = max over valid enc positions; a = pool@W_pool.T + b_reg ----
__global__ void pool_kernel(const float* __restrict__ enc, const int* __restrict__ enc_len,
                            const float* __restrict__ W_pool, const float* __restrict__ b_reg,
                            float* __restrict__ a_ws) {
    __shared__ float pool_s[HH];
    const int b = blockIdx.x, c = threadIdx.x;
    const int len = enc_len[b];
    float mx = -INFINITY;
    for (int t = 0; t < len; ++t)
        mx = fmaxf(mx, enc[(b * TENC + t) * HH + c]);
    pool_s[c] = mx;
    __syncthreads();
    if (c < 16) {
        float d = 0.0f;
        if (c < NCLS) {
            d = b_reg[c];
            for (int h = 0; h < HH; ++h) d += pool_s[h] * W_pool[c * HH + h];
        }
        a_ws[b * 16 + c] = d;
    }
}

// ---------------- one fused GRU decoder timestep ----------------
__global__ __launch_bounds__(256)
void step_kernel(const float* __restrict__ h_in, float* __restrict__ h_out,
                 const float* __restrict__ feat, const float* __restrict__ labels,
                 const float* __restrict__ mask, const float* __restrict__ emb,
                 const float* __restrict__ b_ih, const float* __restrict__ b_hh,
                 const float* __restrict__ a_ws,
                 const unsigned short* __restrict__ Wih, const unsigned short* __restrict__ Whh,
                 const unsigned short* __restrict__ Wrg,
                 float* __restrict__ out, int t) {
    __shared__ __align__(16) unsigned short h_bf[BTILE * HH];    // 16 KB
    __shared__ __align__(16) unsigned short x_bf[BTILE * KIH];   //  8 KB
    __shared__ __align__(16) float          h_f [BTILE * HH];    // 32 KB
    __shared__ __align__(16) float          bmiL[BTILE * 16];    //  2 KB

    const int tid  = threadIdx.x;
    const int lane = tid & 31;
    const int wid  = tid >> 5;
    const int b0   = blockIdx.x * BTILE;

    // ---- phase 1: stage h (f32 + bf16) and x_t (bf16) into LDS ----
    for (int i = tid; i < BTILE * HH; i += 256) {
        float v = h_in[b0 * HH + i];
        h_f[i]  = v;
        h_bf[i] = f2bf(v);
    }
    for (int i = tid; i < BTILE * KIH; i += 256) {
        int m = i >> 7, k = i & (KIH - 1);
        float v = (k < EE) ? feat[((b0 + m) * TDEC + t) * EE + k] : 0.0f;
        x_bf[i] = f2bf(v);
    }
    __syncthreads();

    // ---- phase 2: bmi_h = h @ W_reg^T + (a + b_reg)   (waves 0..1, one M-tile each) ----
    if (wid < (BTILE / 16)) {
        const int mt = wid;
        v8f acc = {};
#pragma unroll
        for (int kb = 0; kb < KHH; kb += 32) {
            v16bf a = load_a_frag(h_bf, HH, mt * 16, kb, lane);
            v16bf b = load_b_frag(Wrg, KHH, 0, kb, lane);
            acc = wmma_bf16(a, b, acc);
        }
        FAcc A; A.v = acc;
        const int n  = lane & 15;
        const int mb = mt * 16 + ((lane >> 4) << 3);
#pragma unroll
        for (int v = 0; v < 8; ++v) {
            int m = mb + v;
            bmiL[m * 16 + n] = A.f[v] + a_ws[(b0 + m) * 16 + n];
        }
    }
    __syncthreads();

    // ---- phase 3: softmax / argmax / embedding mix, fill x_bf[:,52:104] (one thread per row) ----
    if (tid < BTILE) {
        const int m = tid, b = b0 + m;
        float* row = &bmiL[m * 16];
        float mx = row[0]; int arg = 0;
#pragma unroll
        for (int k = 1; k < NCLS; ++k)
            if (row[k] > mx) { mx = row[k]; arg = k; }
        float e[NCLS], s = 0.0f;
#pragma unroll
        for (int k = 0; k < NCLS; ++k) { e[k] = __expf(row[k] - mx); s += e[k]; }
        const float inv = 1.0f / s;
        const int idx = t * BB + b;
        float* out_lbl  = out;
        float* out_prob = out + TDEC * BB;
        float* out_disc = out + TDEC * BB + TDEC * BB * NCLS;
        float* out_kl   = out + TDEC * BB + 2 * TDEC * BB * NCLS;
#pragma unroll
        for (int k = 0; k < NCLS; ++k) {
            out_disc[idx * NCLS + k] = row[k];
            out_prob[idx * NCLS + k] = e[k] * inv;
        }
        const int code = kLabelCodes[arg];
        out_lbl[idx] = (float)arg;
        out_kl[idx]  = (float)code;
        const float mt_ = mask[b * TDEC + t];
        for (int j = 0; j < EE; ++j) {
            float lab = labels[(b * TDEC + t) * EE + j];
            float be  = emb[code * EE + j];
            x_bf[m * KIH + EE + j] = f2bf(mt_ * lab + (1.0f - mt_) * be);
        }
    }
    __syncthreads();

    // ---- phase 4: gate GEMMs (gi = x@W_ih^T, gh = h@W_hh^T) + fused GRU elementwise ----
    // units: (BTILE/16) M-tiles x 16 column tiles of H ; each unit keeps r/z/n tiles of gi & gh in regs
    const int NUNITS = (BTILE / 16) * 16;
    for (int u = wid; u < NUNITS; u += 8) {
        const int mt = u >> 4, ct = u & 15;
        v8f ir = {}, iz = {}, inn = {}, hr = {}, hz = {}, hn = {};
#pragma unroll
        for (int kb = 0; kb < KIH; kb += 32) {
            v16bf a = load_a_frag(x_bf, KIH, mt * 16, kb, lane);
            ir  = wmma_bf16(a, load_b_frag(Wih, KIH, ct * 16,            kb, lane), ir);
            iz  = wmma_bf16(a, load_b_frag(Wih, KIH, HH + ct * 16,       kb, lane), iz);
            inn = wmma_bf16(a, load_b_frag(Wih, KIH, 2 * HH + ct * 16,   kb, lane), inn);
        }
#pragma unroll
        for (int kb = 0; kb < KHH; kb += 32) {
            v16bf a = load_a_frag(h_bf, HH, mt * 16, kb, lane);
            hr = wmma_bf16(a, load_b_frag(Whh, KHH, ct * 16,           kb, lane), hr);
            hz = wmma_bf16(a, load_b_frag(Whh, KHH, HH + ct * 16,      kb, lane), hz);
            hn = wmma_bf16(a, load_b_frag(Whh, KHH, 2 * HH + ct * 16,  kb, lane), hn);
        }
        FAcc IR, IZ, IN, HR, HZ, HN;
        IR.v = ir; IZ.v = iz; IN.v = inn; HR.v = hr; HZ.v = hz; HN.v = hn;
        const int n_ = lane & 15;
        const int mb = mt * 16 + ((lane >> 4) << 3);
        const int cr = ct * 16 + n_;
#pragma unroll
        for (int v = 0; v < 8; ++v) {
            const int m = mb + v, b = b0 + m;
            float r  = sigmoidf_(IR.f[v] + b_ih[cr]          + HR.f[v] + b_hh[cr]);
            float z  = sigmoidf_(IZ.f[v] + b_ih[HH + cr]     + HZ.f[v] + b_hh[HH + cr]);
            float nn = tanhf(IN.f[v] + b_ih[2 * HH + cr] + r * (HN.f[v] + b_hh[2 * HH + cr]));
            float ho = h_f[m * HH + cr];
            h_out[b * HH + cr] = (1.0f - z) * nn + z * ho;
        }
    }
}

extern "C" void kernel_launch(void* const* d_in, const int* in_sizes, int n_in,
                              void* d_out, int out_size, void* d_ws, size_t ws_size,
                              hipStream_t stream) {
    const float* feat   = (const float*)d_in[0];
    const float* labels = (const float*)d_in[1];
    const float* enc    = (const float*)d_in[2];
    const float* h_n    = (const float*)d_in[3];
    const float* mask   = (const float*)d_in[4];
    const int*   enclen = (const int*)  d_in[5];
    const float* emb    = (const float*)d_in[6];
    const float* W_reg  = (const float*)d_in[7];
    const float* b_reg  = (const float*)d_in[8];
    const float* W_pool = (const float*)d_in[9];
    const float* W_ih   = (const float*)d_in[10];
    const float* W_hh   = (const float*)d_in[11];
    const float* b_ih   = (const float*)d_in[12];
    const float* b_hh   = (const float*)d_in[13];

    char* ws = (char*)d_ws;
    float*          a_ws = (float*)(ws + OFF_A);
    float*          hb0  = (float*)(ws + OFF_H0);
    float*          hb1  = (float*)(ws + OFF_H1);
    unsigned short* Wih  = (unsigned short*)(ws + OFF_WIH);
    unsigned short* Whh  = (unsigned short*)(ws + OFF_WHH);
    unsigned short* Wrg  = (unsigned short*)(ws + OFF_WRG);

    const int ncv = (G3 * KIH + G3 * KHH + 16 * KHH + 255) / 256;
    convert_weights<<<ncv, 256, 0, stream>>>(W_ih, W_hh, W_reg, Wih, Whh, Wrg);
    pool_kernel<<<BB, 256, 0, stream>>>(enc, enclen, W_pool, b_reg, a_ws);

    float* hbuf[2] = {hb0, hb1};
    for (int t = 0; t < TDEC; ++t) {
        const float* hin = (t == 0) ? h_n : hbuf[(t - 1) & 1];
        float*       hout = hbuf[t & 1];
        step_kernel<<<BB / BTILE, 256, 0, stream>>>(hin, hout, feat, labels, mask, emb,
                                                    b_ih, b_hh, a_ws, Wih, Whh, Wrg,
                                                    (float*)d_out, t);
    }
}
